// Net_31576599560244
// MI455X (gfx1250) — matrix operand order repaired and checked by reading, not compiled, not run on hardware
//
#include <hip/hip_runtime.h>
#include <hip/hip_bf16.h>

// ---------------------------------------------------------------------------
// Mamba-2-layer network for MI455X (gfx1250, wave32, WMMA).
// GEMMs: bf16 WMMA (V_WMMA_F32_16X16X32_BF16, f32 accum).
//   - B pre-packed into WMMA fragment layout -> 2x global_load_b128 / lane
//   - A tile staged into LDS with GLOBAL_LOAD_ASYNC_TO_LDS_B128 (ASYNCcnt),
//     fragments read back as 2x ds_load_b128 / lane
//   - two M-tiles per wave share each B fragment
// Selective scan is a 3-pass chunked parallel scan.
// ---------------------------------------------------------------------------

#define L_SEQ   16384
#define D_MODEL 256
#define D_INNER 512
#define D_STATE 16
#define DT_RANK 16
#define N_CHUNK 128
#define L_CHUNK (L_SEQ / N_CHUNK)
#define EPSF    1e-5f

typedef __attribute__((ext_vector_type(16))) __bf16        v16bf;
typedef __attribute__((ext_vector_type(8)))  float         v8f;
typedef __attribute__((ext_vector_type(4)))  unsigned int  u32x4;

union FragBF { unsigned int u[8]; u32x4 q[2]; v16bf v; };

__device__ __forceinline__ float gelu_tanh(float x) {
    float x3 = x * x * x;
    return 0.5f * x * (1.0f + tanhf(0.7978845608028654f * (x + 0.044715f * x3)));
}
__device__ __forceinline__ float silu(float x) {
    return x / (1.0f + expf(-x));
}

// ---------------------------------------------------------------------------
// Pack f32 weight B[K,N] (row-major) into the WMMA B-fragment dword layout:
//   out[((nt*Ksteps + ks)*32 + lane)*8 + j] = {B[k][n], B[k+1][n]} (bf16x2)
//   n = nt*16 + (lane&15),  k = ks*32 + (j>>2)*16 + (lane>>4)*8 + 2*(j&3)
// GEMM then reads each lane's 8 dwords as two contiguous b128 loads.
// ---------------------------------------------------------------------------
__global__ void pack_b_kernel(const float* __restrict__ B, unsigned int* __restrict__ out,
                              int K, int N)
{
    int id = blockIdx.x * blockDim.x + threadIdx.x;
    int Ksteps = K >> 5;
    int total = Ksteps * (N >> 4) * 256;
    if (id >= total) return;
    int j  = id & 7;
    int l  = (id >> 3) & 31;
    int ks = (id >> 8) % Ksteps;
    int nt = id / (Ksteps << 8);
    int n  = nt * 16 + (l & 15);
    int k  = ks * 32 + (j >> 2) * 16 + (l >> 4) * 8 + 2 * (j & 3);
    __bf16 lo = (__bf16)B[k * N + n];
    __bf16 hi = (__bf16)B[(k + 1) * N + n];
    out[id] = (unsigned int)__builtin_bit_cast(unsigned short, lo) |
              ((unsigned int)__builtin_bit_cast(unsigned short, hi) << 16);
}

// ---------------------------------------------------------------------------
// bf16 WMMA GEMM:  C[M,N](f32) = A[M,K](bf16,row) * B(packed)
// Block covers 32 rows x (16*WN) cols; each wave: two 16x16 tiles along M.
// grid = (M/32, N/(16*WN)), block = 32*WN.
// accumulate!=0 -> C += result (fuses the residual add).
// ---------------------------------------------------------------------------
template<int WN>
__launch_bounds__(32 * WN)
__global__ void gemm_bf16_wmma(const __bf16* __restrict__ A,
                               const unsigned int* __restrict__ Bp,
                               float* __restrict__ C,
                               int M, int N, int K, int accumulate)
{
    __shared__ __align__(16) unsigned int lds[512];   // A tile: 32 rows x 32 k (bf16)

    const int tid    = threadIdx.x;
    const int lane   = tid & 31;
    const int tileM  = blockIdx.x * 32;
    const int tileNt = blockIdx.y * WN + (tid >> 5);
    const int tileN  = tileNt * 16;
    const int hf     = lane >> 4;
    const int idx    = lane & 15;
    const int Ksteps = K >> 5;

    v8f acc0 = {}, acc1 = {};
    const u32x4* Aq = (const u32x4*)A;        // 8 bf16 per element
    const u32x4* Bq = (const u32x4*)Bp;
    const u32x4* lq = (const u32x4*)lds;
    // LDS byte offset of the staging buffer (flat aperture: low 32 bits)
    const unsigned ldsBase = (unsigned)(size_t)(void*)lds;
    const int rowQ = K >> 3;                  // row stride in u32x4 units

    for (int ks = 0; ks < Ksteps; ++ks) {
        const int k0 = ks << 5;
        // ---- stage A tile (32 rows x 32 k = 128 x 16B) via async copy to LDS
        for (int i = tid; i < 128; i += 32 * WN) {
            const u32x4* gp = Aq + (size_t)(tileM + (i >> 2)) * rowQ + (k0 >> 3) + (i & 3);
            unsigned dst = ldsBase + i * 16;
            asm volatile("global_load_async_to_lds_b128 %0, %1, off"
                         :: "v"(dst), "v"(gp) : "memory");
        }
        if (k0 + 32 < K)   // prefetch next A k-tile (global_prefetch_b8)
            __builtin_prefetch(Aq + (size_t)(tileM + (tid & 31)) * rowQ + ((k0 + 32) >> 3), 0, 1);
        asm volatile("s_wait_asynccnt 0x0" ::: "memory");
        __syncthreads();

        // ---- A fragments: u[0..3]/u[4..7] are contiguous dwords in LDS
        FragBF fa0, fa1, fb;
        fa0.q[0] = lq[idx * 4 + hf];
        fa0.q[1] = lq[idx * 4 + 2 + hf];
        fa1.q[0] = lq[(idx + 16) * 4 + hf];
        fa1.q[1] = lq[(idx + 16) * 4 + 2 + hf];
        // ---- B fragment: pre-packed, 2x b128 per lane, coalesced across wave
        int bi = ((tileNt * Ksteps + ks) * 32 + lane) * 2;
        fb.q[0] = Bq[bi];
        fb.q[1] = Bq[bi + 1];

        acc0 = __builtin_amdgcn_wmma_f32_16x16x32_bf16(
                   false, fa0.v, false, fb.v, (short)0, acc0, false, false);
        acc1 = __builtin_amdgcn_wmma_f32_16x16x32_bf16(
                   false, fa1.v, false, fb.v, (short)0, acc1, false, false);
        __syncthreads();
    }

#pragma unroll
    for (int v = 0; v < 8; ++v) {
        int r0  = tileM + v + 8 * hf;
        int r1  = r0 + 16;
        int col = tileN + idx;
        if (accumulate) { C[r0 * N + col] += acc0[v]; C[r1 * N + col] += acc1[v]; }
        else            { C[r0 * N + col]  = acc0[v]; C[r1 * N + col]  = acc1[v]; }
    }
}

// ---------------------------------------------------------------------------
// embed: h = gelu(LN(x @ W_in + b_in)).  block = 256 (one row / block).
// ---------------------------------------------------------------------------
__global__ void embed_kernel(const float* __restrict__ x, const float* __restrict__ Win,
                             const float* __restrict__ bin, const float* __restrict__ g,
                             const float* __restrict__ b, float* __restrict__ h)
{
    int t = blockIdx.x, j = threadIdx.x;
    float acc = bin[j];
#pragma unroll
    for (int k = 0; k < 32; ++k) acc += x[t * 32 + k] * Win[k * 256 + j];

    __shared__ float red[256];
    red[j] = acc; __syncthreads();
    for (int s = 128; s > 0; s >>= 1) { if (j < s) red[j] += red[j + s]; __syncthreads(); }
    float mean = red[0] * (1.0f / 256.0f); __syncthreads();
    float d = acc - mean;
    red[j] = d * d; __syncthreads();
    for (int s = 128; s > 0; s >>= 1) { if (j < s) red[j] += red[j + s]; __syncthreads(); }
    float var = red[0] * (1.0f / 256.0f);
    float v = d * rsqrtf(var + EPSF) * g[j] + b[j];
    h[t * 256 + j] = gelu_tanh(v);
}

// ---------------------------------------------------------------------------
// RMSNorm -> bf16 (block = 256, one row / block)
// ---------------------------------------------------------------------------
__global__ void rms_bf16_kernel(const float* __restrict__ h, const float* __restrict__ w,
                                __bf16* __restrict__ o)
{
    int t = blockIdx.x, j = threadIdx.x;
    float v = h[t * 256 + j];
    __shared__ float red[256];
    red[j] = v * v; __syncthreads();
    for (int s = 128; s > 0; s >>= 1) { if (j < s) red[j] += red[j + s]; __syncthreads(); }
    float ms = red[0] * (1.0f / 256.0f);
    o[t * 256 + j] = (__bf16)(v * rsqrtf(ms + EPSF) * w[j]);
}

// ---------------------------------------------------------------------------
// LayerNorm -> bf16 (block = D, one row / block, D power of two <= 256)
// ---------------------------------------------------------------------------
__global__ void ln_bf16_kernel(const float* __restrict__ h, const float* __restrict__ g,
                               const float* __restrict__ b, __bf16* __restrict__ o, int D)
{
    int t = blockIdx.x, j = threadIdx.x;
    float v = h[t * D + j];
    __shared__ float red[256];
    red[j] = v; __syncthreads();
    for (int s = D >> 1; s > 0; s >>= 1) { if (j < s) red[j] += red[j + s]; __syncthreads(); }
    float mean = red[0] / (float)D; __syncthreads();
    float d = v - mean;
    red[j] = d * d; __syncthreads();
    for (int s = D >> 1; s > 0; s >>= 1) { if (j < s) red[j] += red[j + s]; __syncthreads(); }
    float var = red[0] / (float)D;
    o[t * D + j] = (__bf16)(d * rsqrtf(var + EPSF) * g[j] + b[j]);
}

// ---------------------------------------------------------------------------
// depthwise causal conv (4 taps) + silu.   xz: [L,1024], xin = xz[:, :512]
// ---------------------------------------------------------------------------
__global__ void conv_silu_kernel(const float* __restrict__ xz, const float* __restrict__ cw,
                                 const float* __restrict__ cb, float* __restrict__ xs,
                                 __bf16* __restrict__ xsb)
{
    int id = blockIdx.x * blockDim.x + threadIdx.x;
    if (id >= L_SEQ * D_INNER) return;
    int t = id >> 9, e = id & 511;
    float acc = cb[e];
#pragma unroll
    for (int i = 0; i < 4; ++i) {
        int tt = t - 3 + i;
        if (tt >= 0) acc += cw[e * 4 + i] * xz[tt * 1024 + e];
    }
    float s = silu(acc);
    xs[id]  = s;
    xsb[id] = (__bf16)s;
}

// ---------------------------------------------------------------------------
// delta = softplus(dbc[:, :16] @ dtW + dtB)
// ---------------------------------------------------------------------------
__global__ void delta_kernel(const float* __restrict__ dbc, const float* __restrict__ dtW,
                             const float* __restrict__ dtB, float* __restrict__ delta)
{
    int id = blockIdx.x * blockDim.x + threadIdx.x;
    if (id >= L_SEQ * D_INNER) return;
    int t = id >> 9, e = id & 511;
    float acc = dtB[e];
#pragma unroll
    for (int r = 0; r < DT_RANK; ++r) acc += dbc[t * 48 + r] * dtW[r * 512 + e];
    delta[id] = (acc > 20.0f) ? acc : log1pf(expf(acc));
}

// ---------------------------------------------------------------------------
// Chunked selective scan.
// Pass A: per (chunk,e): local scan from h=0; store chunk delta-sum S and
//         local end-state Hloc.  (prod of dA over chunk == exp(A * sum(delta)))
// ---------------------------------------------------------------------------
__global__ void scanA_kernel(const float* __restrict__ delta, const float* __restrict__ xs,
                             const float* __restrict__ dbc, const float* __restrict__ A_log,
                             float* __restrict__ S, float* __restrict__ Hloc)
{
    int c = blockIdx.x;
    int e = blockIdx.y * blockDim.x + threadIdx.x;
    float Ae[D_STATE];
#pragma unroll
    for (int n = 0; n < D_STATE; ++n) Ae[n] = -expf(A_log[e * D_STATE + n]);
    float hs[D_STATE];
#pragma unroll
    for (int n = 0; n < D_STATE; ++n) hs[n] = 0.0f;
    float ds = 0.0f;
    for (int tt = 0; tt < L_CHUNK; ++tt) {
        int t = c * L_CHUNK + tt;
        float d  = delta[t * 512 + e];
        float dx = d * xs[t * 512 + e];
        ds += d;
#pragma unroll
        for (int n = 0; n < D_STATE; ++n)
            hs[n] = expf(d * Ae[n]) * hs[n] + dx * dbc[t * 48 + 16 + n];
    }
    S[c * 512 + e] = ds;
#pragma unroll
    for (int n = 0; n < D_STATE; ++n) Hloc[(c * 512 + e) * D_STATE + n] = hs[n];
}

// Pass B: serial carry across 128 chunks for each of 8192 (e,n) lanes.
__global__ void scanB_kernel(const float* __restrict__ A_log, const float* __restrict__ S,
                             const float* __restrict__ Hloc, float* __restrict__ Hin)
{
    int id = blockIdx.x * blockDim.x + threadIdx.x;   // 0..8191
    if (id >= D_INNER * D_STATE) return;
    int e = id >> 4, n = id & 15;
    float Ae  = -expf(A_log[id]);
    float hin = 0.0f;
    for (int c = 0; c < N_CHUNK; ++c) {
        Hin[(c * 512 + e) * D_STATE + n] = hin;
        hin = expf(Ae * S[c * 512 + e]) * hin + Hloc[(c * 512 + e) * D_STATE + n];
    }
}

// Pass C: replay chunk from correct incoming state; fuse y*silu(z) -> bf16.
__global__ void scanC_kernel(const float* __restrict__ delta, const float* __restrict__ xs,
                             const float* __restrict__ dbc, const float* __restrict__ A_log,
                             const float* __restrict__ Dp, const float* __restrict__ xz,
                             const float* __restrict__ Hin, __bf16* __restrict__ ybf)
{
    int c = blockIdx.x;
    int e = blockIdx.y * blockDim.x + threadIdx.x;
    float Ae[D_STATE];
#pragma unroll
    for (int n = 0; n < D_STATE; ++n) Ae[n] = -expf(A_log[e * D_STATE + n]);
    float hs[D_STATE];
#pragma unroll
    for (int n = 0; n < D_STATE; ++n) hs[n] = Hin[(c * 512 + e) * D_STATE + n];
    float Dv = Dp[e];
    for (int tt = 0; tt < L_CHUNK; ++tt) {
        int t = c * L_CHUNK + tt;
        float d  = delta[t * 512 + e];
        float dx = d * xs[t * 512 + e];
        float y  = xs[t * 512 + e] * Dv;
#pragma unroll
        for (int n = 0; n < D_STATE; ++n) {
            hs[n] = expf(d * Ae[n]) * hs[n] + dx * dbc[t * 48 + 16 + n];
            y    += hs[n] * dbc[t * 48 + 32 + n];
        }
        float z = xz[t * 1024 + 512 + e];
        ybf[t * 512 + e] = (__bf16)(y * silu(z));
    }
}

// ---------------------------------------------------------------------------
// head: v = gelu(mlp + b1); LN(ln3); out = tanh(v @ W2 + b2).  block = 128.
// ---------------------------------------------------------------------------
__global__ void head_kernel(const float* __restrict__ mlp, const float* __restrict__ b1,
                            const float* __restrict__ g3, const float* __restrict__ b3,
                            const float* __restrict__ W2, const float* __restrict__ b2,
                            float* __restrict__ out)
{
    int t = blockIdx.x, j = threadIdx.x;
    float v = gelu_tanh(mlp[t * 128 + j] + b1[j]);
    __shared__ float red[128];
    red[j] = v; __syncthreads();
    for (int s = 64; s > 0; s >>= 1) { if (j < s) red[j] += red[j + s]; __syncthreads(); }
    float mean = red[0] * (1.0f / 128.0f); __syncthreads();
    float d = v - mean;
    red[j] = d * d; __syncthreads();
    for (int s = 64; s > 0; s >>= 1) { if (j < s) red[j] += red[j + s]; __syncthreads(); }
    float var = red[0] * (1.0f / 128.0f); __syncthreads();
    float hn = d * rsqrtf(var + EPSF) * g3[j] + b3[j];
    red[j] = hn * W2[j]; __syncthreads();
    for (int s = 64; s > 0; s >>= 1) { if (j < s) red[j] += red[j + s]; __syncthreads(); }
    if (j == 0) out[t] = tanhf(red[0] + b2[0]);
}

// ---------------------------------------------------------------------------
extern "C" void kernel_launch(void* const* d_in, const int* in_sizes, int n_in,
                              void* d_out, int out_size, void* d_ws, size_t ws_size,
                              hipStream_t stream)
{
    const float* x     = (const float*)d_in[0];
    const float* W_in  = (const float*)d_in[1];
    const float* b_in  = (const float*)d_in[2];
    const float* ln1_g = (const float*)d_in[3];
    const float* ln1_b = (const float*)d_in[4];
    const float* rms_w = (const float*)d_in[5];
    const float* inW   = (const float*)d_in[6];
    const float* convW = (const float*)d_in[7];
    const float* convB = (const float*)d_in[8];
    const float* xpW   = (const float*)d_in[9];
    const float* dtW   = (const float*)d_in[10];
    const float* dtB   = (const float*)d_in[11];
    const float* A_log = (const float*)d_in[12];
    const float* Dp    = (const float*)d_in[13];
    const float* outW  = (const float*)d_in[14];
    const float* ln2_g = (const float*)d_in[15];
    const float* ln2_b = (const float*)d_in[16];
    const float* W1    = (const float*)d_in[17];
    const float* b1    = (const float*)d_in[18];
    const float* ln3_g = (const float*)d_in[19];
    const float* ln3_b = (const float*)d_in[20];
    const float* W2    = (const float*)d_in[21];
    const float* b2    = (const float*)d_in[22];

    // -------- workspace layout --------
    char* p = (char*)d_ws;
    auto alloc = [&](size_t bytes) -> void* {
        void* r = (void*)p;
        p += (bytes + 255) & ~(size_t)255;
        return r;
    };
    float*  h     = (float*)alloc((size_t)L_SEQ * 256 * 4);
    float*  xz    = (float*)alloc((size_t)L_SEQ * 1024 * 4);
    float*  xs    = (float*)alloc((size_t)L_SEQ * 512 * 4);
    float*  delta = (float*)alloc((size_t)L_SEQ * 512 * 4);
    float*  dbc   = (float*)alloc((size_t)L_SEQ * 48 * 4);
    float*  Ssum  = (float*)alloc((size_t)N_CHUNK * 512 * 4);
    float*  Hloc  = (float*)alloc((size_t)N_CHUNK * 512 * 16 * 4);
    float*  Hin   = (float*)alloc((size_t)N_CHUNK * 512 * 16 * 4);
    float*  mlp   = (float*)alloc((size_t)L_SEQ * 128 * 4);
    __bf16* hn    = (__bf16*)alloc((size_t)L_SEQ * 256 * 2);
    __bf16* xsb   = (__bf16*)alloc((size_t)L_SEQ * 512 * 2);
    __bf16* ybf   = (__bf16*)alloc((size_t)L_SEQ * 512 * 2);
    __bf16* hn2   = (__bf16*)alloc((size_t)L_SEQ * 256 * 2);
    unsigned int* wInB  = (unsigned int*)alloc((size_t)256 * 1024 * 2);  // packed bf16x2
    unsigned int* wXpB  = (unsigned int*)alloc((size_t)512 * 48 * 2);
    unsigned int* wOutB = (unsigned int*)alloc((size_t)512 * 256 * 2);
    unsigned int* wW1B  = (unsigned int*)alloc((size_t)256 * 128 * 2);

    const int EL = L_SEQ * D_INNER;

    // -------- stem --------
    embed_kernel<<<L_SEQ, 256, 0, stream>>>(x, W_in, b_in, ln1_g, ln1_b, h);

    // -------- mamba layers --------
    for (int l = 0; l < 2; ++l) {
        const float* inW_l   = inW   + (size_t)l * 256 * 1024;
        const float* convW_l = convW + (size_t)l * 512 * 4;
        const float* convB_l = convB + (size_t)l * 512;
        const float* xpW_l   = xpW   + (size_t)l * 512 * 48;
        const float* dtW_l   = dtW   + (size_t)l * 16 * 512;
        const float* dtB_l   = dtB   + (size_t)l * 512;
        const float* A_l     = A_log + (size_t)l * 512 * 16;
        const float* Dp_l    = Dp    + (size_t)l * 512;
        const float* outW_l  = outW  + (size_t)l * 512 * 256;
        const float* rms_l   = rms_w + (size_t)l * 256;

        rms_bf16_kernel<<<L_SEQ, 256, 0, stream>>>(h, rms_l, hn);

        pack_b_kernel<<<(256 * 1024 / 2 + 255) / 256, 256, 0, stream>>>(inW_l, wInB, 256, 1024);
        gemm_bf16_wmma<8><<<dim3(L_SEQ / 32, 1024 / 128), 256, 0, stream>>>(
            hn, wInB, xz, L_SEQ, 1024, 256, 0);

        conv_silu_kernel<<<(EL + 255) / 256, 256, 0, stream>>>(xz, convW_l, convB_l, xs, xsb);

        pack_b_kernel<<<(512 * 48 / 2 + 255) / 256, 256, 0, stream>>>(xpW_l, wXpB, 512, 48);
        gemm_bf16_wmma<3><<<dim3(L_SEQ / 32, 1), 96, 0, stream>>>(
            xsb, wXpB, dbc, L_SEQ, 48, 512, 0);

        delta_kernel<<<(EL + 255) / 256, 256, 0, stream>>>(dbc, dtW_l, dtB_l, delta);

        scanA_kernel<<<dim3(N_CHUNK, D_INNER / 128), 128, 0, stream>>>(
            delta, xs, dbc, A_l, Ssum, Hloc);
        scanB_kernel<<<(D_INNER * D_STATE + 255) / 256, 256, 0, stream>>>(
            A_l, Ssum, Hloc, Hin);
        scanC_kernel<<<dim3(N_CHUNK, D_INNER / 128), 128, 0, stream>>>(
            delta, xs, dbc, A_l, Dp_l, xz, Hin, ybf);

        pack_b_kernel<<<(512 * 256 / 2 + 255) / 256, 256, 0, stream>>>(outW_l, wOutB, 512, 256);
        // accumulate=1 fuses the residual: h += ybf @ outW
        gemm_bf16_wmma<8><<<dim3(L_SEQ / 32, 256 / 128), 256, 0, stream>>>(
            ybf, wOutB, h, L_SEQ, 256, 512, 1);
    }

    // -------- head --------
    ln_bf16_kernel<<<L_SEQ, 256, 0, stream>>>(h, ln2_g, ln2_b, hn2, 256);
    pack_b_kernel<<<(256 * 128 / 2 + 255) / 256, 256, 0, stream>>>(W1, wW1B, 256, 128);
    gemm_bf16_wmma<8><<<dim3(L_SEQ / 32, 1), 256, 0, stream>>>(
        hn2, wW1B, mlp, L_SEQ, 128, 256, 0);
    head_kernel<<<L_SEQ, 128, 0, stream>>>(mlp, b1, ln3_g, ln3_b, W2, b2, (float*)d_out);
}
